// EfficientTransformerBlock_13580686590104
// MI455X (gfx1250) — matrix operand order repaired and checked by reading, not compiled, not run on hardware
//
#include <hip/hip_runtime.h>
#include <hip/hip_bf16.h>
#include <math.h>

// Problem constants (match reference)
#define Bb   2
#define Nseq 2048
#define Cc   1024
#define Hh   16
#define Dd   64
#define FFf  4096
#define Mrows (Bb*Nseq)               // 4096
#define BNC  ((size_t)Bb*Nseq*Cc)     // 4,194,304 elements

// Flip to 0 if the async-to-LDS mnemonic is rejected by the assembler.
#define USE_ASYNC_LDS 1

typedef _Float16 half_t;
typedef __attribute__((ext_vector_type(16))) _Float16 v16h;
typedef __attribute__((ext_vector_type(8)))  float    v8f;

// ---------------------------------------------------------------- async copy
// 16B global -> LDS, tracked by ASYNCcnt (CDNA5 GLOBAL_LOAD_ASYNC_TO_LDS_B128).
// VDST carries the LDS byte address = low 32 bits of the flat shared pointer.
__device__ __forceinline__ void cp_async16(half_t* dst_lds, const half_t* src) {
#if USE_ASYNC_LDS
  unsigned int l = (unsigned int)(uintptr_t)dst_lds;
  asm volatile("global_load_async_to_lds_b128 %0, %1, off"
               :: "v"(l), "v"(src) : "memory");
#else
  *(uint4*)dst_lds = *(const uint4*)src;
#endif
}
__device__ __forceinline__ void async_wait() {
#if USE_ASYNC_LDS
  asm volatile("s_wait_asynccnt 0x0" ::: "memory");
#endif
}

// ---------------------------------------------------------------- WMMA core
__device__ __forceinline__ v8f wmma32(v16h a, v16h b, v8f c) {
  return __builtin_amdgcn_wmma_f32_16x16x32_f16(
      false, a, false, b, (short)0, c, false, false);
}

// A fragment: 16x32 f16, row-major source, leading dim ld (halves).
__device__ __forceinline__ v16h frag_a(const half_t* __restrict__ src, int ld) {
  int lane = threadIdx.x & 31;
  int m = lane & 15, hi = lane >> 4;
  const half_t* p = src + m * ld;
  v16h f;
#pragma unroll
  for (int i = 0; i < 8; i++) f[i]     = p[hi * 16 + i];
#pragma unroll
  for (int i = 0; i < 8; i++) f[8 + i] = p[hi * 16 + 8 + i];
  return f;
}

// B fragment: 32x16 f16 from n-major source: src[n*ld + k], k contiguous per lane.
__device__ __forceinline__ v16h frag_b(const half_t* __restrict__ src, int ld) {
  int lane = threadIdx.x & 31;
  int n = lane & 15, kb = (lane >> 4) * 16;
  const half_t* p = src + n * ld + kb;
  v16h f;
#pragma unroll
  for (int i = 0; i < 16; i++) f[i] = p[i];
  return f;
}

// ---------------------------------------------------------------- f32 -> f16
__global__ __launch_bounds__(256) void cvt_f16(const float* __restrict__ in,
                                               half_t* __restrict__ out, size_t n) {
  size_t i = (size_t)blockIdx.x * 256 + threadIdx.x;
  size_t stride = (size_t)gridDim.x * 256;
  for (; i < n; i += stride) out[i] = (half_t)in[i];
}

// ---------------------------------------------------------------- LayerNorm (row = 1024)
__global__ __launch_bounds__(256) void layernorm_f16(const float* __restrict__ x,
                                                     const float* __restrict__ g,
                                                     const float* __restrict__ b,
                                                     half_t* __restrict__ out) {
  __shared__ float rs[8], rs2[8];
  int row = blockIdx.x;
  const float* xr = x + (size_t)row * Cc;
  float4 v = ((const float4*)xr)[threadIdx.x];
  float s  = v.x + v.y + v.z + v.w;
  float s2 = v.x * v.x + v.y * v.y + v.z * v.z + v.w * v.w;
#pragma unroll
  for (int off = 16; off >= 1; off >>= 1) {
    s  += __shfl_xor(s, off);
    s2 += __shfl_xor(s2, off);
  }
  int wave = threadIdx.x >> 5, lane = threadIdx.x & 31;
  if (lane == 0) { rs[wave] = s; rs2[wave] = s2; }
  __syncthreads();
  if (threadIdx.x == 0) {
    float ts = 0.f, ts2 = 0.f;
#pragma unroll
    for (int i = 0; i < 8; i++) { ts += rs[i]; ts2 += rs2[i]; }
    float mu = ts * (1.0f / Cc);
    float var = ts2 * (1.0f / Cc) - mu * mu;
    rs[0] = mu;
    rs2[0] = rsqrtf(var + 1e-5f);
  }
  __syncthreads();
  float mu = rs[0], rstd = rs2[0];
  float4 gg = ((const float4*)g)[threadIdx.x];
  float4 bb = ((const float4*)b)[threadIdx.x];
  half_t* op = out + (size_t)row * Cc + threadIdx.x * 4;
  op[0] = (half_t)((v.x - mu) * rstd * gg.x + bb.x);
  op[1] = (half_t)((v.y - mu) * rstd * gg.y + bb.y);
  op[2] = (half_t)((v.z - mu) * rstd * gg.z + bb.z);
  op[3] = (half_t)((v.w - mu) * rstd * gg.w + bb.w);
}

// ---------------------------------------------------------------- tiled WMMA GEMM
// C[M,N] = A[M,K] @ W[N,K]^T. Block tile 128x64, double-buffered async LDS pipeline.
// 8 waves (4 M x 2 N), wave tile 32x32 -> 4 WMMA per K-step of 32.
// MODE 0: qkv scatter+scale, 1: bias+GELU->f16, 2: bias+resid->f32
template <int MODE>
__global__ __launch_bounds__(256) void gemm_wmma(
    const half_t* __restrict__ A, const half_t* __restrict__ W,
    const float* __restrict__ bias, const float* __restrict__ resid,
    half_t* __restrict__ out16, float* __restrict__ out32,
    int M, int N, int K) {
  __shared__ half_t As[2][128][40];   // 20 KB  (80B rows: 16B-aligned chunks)
  __shared__ half_t Bs[2][64][40];    // 10 KB
  int tid = threadIdx.x;
  int wave = tid >> 5, lane = tid & 31;
  int wm = wave & 3, wn = wave >> 2;
  int m0 = blockIdx.x * 128, n0 = blockIdx.y * 64;

  auto fill = [&](int k0, int buf) {
    int ra = tid >> 1;                 // 0..127
    int ca = (tid & 1) * 16;           // 0 / 16 halves
    const half_t* ga = &A[(size_t)(m0 + ra) * K + k0 + ca];
    cp_async16(&As[buf][ra][ca], ga);
    cp_async16(&As[buf][ra][ca + 8], ga + 8);
    int rb = tid >> 2;                 // 0..63
    int cb = (tid & 3) * 8;
    cp_async16(&Bs[buf][rb][cb], &W[(size_t)(n0 + rb) * K + k0 + cb]);
  };

  v8f acc[2][2] = {{v8f{}, v8f{}}, {v8f{}, v8f{}}};
  int cur = 0;
  fill(0, 0);
  async_wait();
  __syncthreads();
  for (int k0 = 0; k0 < K; k0 += 32) {
    int nxt = cur ^ 1;
    if (k0 + 32 < K) fill(k0 + 32, nxt);   // overlap with compute below
    v16h a0 = frag_a(&As[cur][wm * 32][0], 40);
    v16h a1 = frag_a(&As[cur][wm * 32 + 16][0], 40);
    v16h b0 = frag_b(&Bs[cur][wn * 32][0], 40);
    v16h b1 = frag_b(&Bs[cur][wn * 32 + 16][0], 40);
    acc[0][0] = wmma32(a0, b0, acc[0][0]);
    acc[0][1] = wmma32(a0, b1, acc[0][1]);
    acc[1][0] = wmma32(a1, b0, acc[1][0]);
    acc[1][1] = wmma32(a1, b1, acc[1][1]);
    async_wait();
    __syncthreads();
    cur = nxt;
  }

  int col = lane & 15, hi = lane >> 4;
#pragma unroll
  for (int s = 0; s < 2; s++) {
#pragma unroll
    for (int t = 0; t < 2; t++) {
      v8f a = acc[s][t];
      int n = n0 + wn * 32 + t * 16 + col;
#pragma unroll
      for (int r = 0; r < 8; r++) {
        int mr = m0 + wm * 32 + s * 16 + hi * 8 + r;
        float val = a[r];
        if constexpr (MODE == 0) {
          int which = n >> 10, rem = n & 1023;
          int head = rem >> 6, d = rem & 63;
          int bb2 = mr >> 11, nn = mr & 2047;
          float sc = (which == 0) ? 0.125f : 1.0f;   // fold 1/sqrt(D) into q
          size_t dst = (size_t)which * BNC +
                       (((size_t)bb2 * Hh + head) * Nseq + nn) * Dd + d;
          out16[dst] = (half_t)(val * sc);
        } else if constexpr (MODE == 1) {
          val += bias[n];
          val = 0.5f * val * (1.0f + erff(val * 0.70710678118f));  // exact GELU
          out16[(size_t)mr * N + n] = (half_t)val;
        } else {
          val += bias[n];
          out32[(size_t)mr * N + n] = resid[(size_t)mr * N + n] + val;
        }
      }
    }
  }
}

// ---------------------------------------------------------------- flash attention
// grid: (Nseq/128, B*H); 256 threads = 8 waves, each wave owns 16 query rows.
__global__ __launch_bounds__(256) void flash_attn(
    const half_t* __restrict__ q, const half_t* __restrict__ k,
    const half_t* __restrict__ v, half_t* __restrict__ ao /* [B,N,C] f16 */) {
  __shared__ half_t Ks[32][Dd];        // 4 KB, row-major (async fill)
  __shared__ half_t Vt[Dd][40];        // 5 KB, V transposed (d-major)
  __shared__ half_t Ps[8][16][40];     // 10 KB, per-wave P staging
  int tid = threadIdx.x;
  int bh = blockIdx.y;
  const half_t* qb = q + (size_t)bh * Nseq * Dd;
  const half_t* kb = k + (size_t)bh * Nseq * Dd;
  const half_t* vb = v + (size_t)bh * Nseq * Dd;
  int wave = tid >> 5, lane = tid & 31;
  int m = lane & 15, hi = lane >> 4;
  int qrow0 = blockIdx.x * 128 + wave * 16;

  // Q fragments (2 K-steps over D=64); q pre-scaled by 1/sqrt(D)
  const half_t* qp = qb + (size_t)(qrow0 + m) * Dd;
  v16h qa[2];
#pragma unroll
  for (int dt = 0; dt < 2; dt++) {
    int d0 = dt * 32;
#pragma unroll
    for (int i = 0; i < 8; i++) qa[dt][i]     = qp[d0 + hi * 16 + i];
#pragma unroll
    for (int i = 0; i < 8; i++) qa[dt][8 + i] = qp[d0 + hi * 16 + 8 + i];
  }

  v8f o[4] = {v8f{}, v8f{}, v8f{}, v8f{}};
  float rm[8], rl[8];
#pragma unroll
  for (int r = 0; r < 8; r++) { rm[r] = -3.0e38f; rl[r] = 0.0f; }

  for (int j0 = 0; j0 < Nseq; j0 += 32) {
    __syncthreads();   // previous iteration's tile reads complete
    {
      int row = tid >> 3, c8 = (tid & 7) * 8;   // 32 rows x 64 halves, 8/thread
      cp_async16(&Ks[row][c8], &kb[(size_t)(j0 + row) * Dd + c8]);
      // V: load row-major, store transposed so P.V B-frags read contiguous
      uint4 vv = *(const uint4*)&vb[(size_t)(j0 + row) * Dd + c8];
      half_t tmp[8];
      *(uint4*)tmp = vv;
#pragma unroll
      for (int i = 0; i < 8; i++) Vt[c8 + i][row] = tmp[i];
    }
    async_wait();
    __syncthreads();

    // S = Q K^T for 32 keys: two 16x16 tiles
    v8f s0 = {}, s1 = {};
#pragma unroll
    for (int dt = 0; dt < 2; dt++) {
      v16h b0 = frag_b(&Ks[0][dt * 32], Dd);
      v16h b1 = frag_b(&Ks[16][dt * 32], Dd);
      s0 = wmma32(qa[dt], b0, s0);
      s1 = wmma32(qa[dt], b1, s1);
    }

    // online softmax; row r lives at VGPR r within each 16-lane half
#pragma unroll
    for (int r = 0; r < 8; r++) {
      float sm = fmaxf(s0[r], s1[r]);
#pragma unroll
      for (int off = 8; off >= 1; off >>= 1) sm = fmaxf(sm, __shfl_xor(sm, off));
      float mn = fmaxf(rm[r], sm);
      float al = __expf(rm[r] - mn);
      float p0 = __expf(s0[r] - mn);
      float p1 = __expf(s1[r] - mn);
      s0[r] = p0; s1[r] = p1;
      float rsum = p0 + p1;
#pragma unroll
      for (int off = 8; off >= 1; off >>= 1) rsum += __shfl_xor(rsum, off);
      rl[r] = rl[r] * al + rsum;
      rm[r] = mn;
#pragma unroll
      for (int dt = 0; dt < 4; dt++) o[dt][r] *= al;
    }

    // stage P (16x32 f16) for WMMA-A reload
    {
      int pcol = lane & 15, rb = hi * 8;
#pragma unroll
      for (int r = 0; r < 8; r++) {
        Ps[wave][rb + r][pcol]      = (half_t)s0[r];
        Ps[wave][rb + r][16 + pcol] = (half_t)s1[r];
      }
    }
    asm volatile("s_wait_dscnt 0x0" ::: "memory");  // intra-wave cross-lane RAW

    v16h pa = frag_a(&Ps[wave][0][0], 40);
#pragma unroll
    for (int dt = 0; dt < 4; dt++) {
      v16h vf = frag_b(&Vt[dt * 16][0], 40);   // contiguous b128 reads
      o[dt] = wmma32(pa, vf, o[dt]);
    }
  }

  // epilogue: divide by l, store into ao[B,N,C]
  int b = bh >> 4, h = bh & 15;
  int col = lane & 15;
#pragma unroll
  for (int dt = 0; dt < 4; dt++) {
#pragma unroll
    for (int r = 0; r < 8; r++) {
      int n = qrow0 + hi * 8 + r;
      int c = h * Dd + dt * 16 + col;
      ao[((size_t)b * Nseq + n) * Cc + c] = (half_t)(o[dt][r] / rl[r]);
    }
  }
}

// ---------------------------------------------------------------- launch
extern "C" void kernel_launch(void* const* d_in, const int* in_sizes, int n_in,
                              void* d_out, int out_size, void* d_ws, size_t ws_size,
                              hipStream_t stream) {
  (void)in_sizes; (void)n_in; (void)out_size; (void)ws_size;
  const float* x      = (const float*)d_in[0];
  const float* ln1_g  = (const float*)d_in[1];
  const float* ln1_b  = (const float*)d_in[2];
  const float* qkv_w  = (const float*)d_in[3];
  const float* proj_w = (const float*)d_in[4];
  const float* proj_b = (const float*)d_in[5];
  const float* ln2_g  = (const float*)d_in[6];
  const float* ln2_b  = (const float*)d_in[7];
  const float* w1     = (const float*)d_in[8];
  const float* b1     = (const float*)d_in[9];
  const float* w2     = (const float*)d_in[10];
  const float* b2     = (const float*)d_in[11];
  float* out = (float*)d_out;

  // workspace carve-up (~120 MiB total)
  char* ws = (char*)d_ws;
  size_t p = 0;
  half_t* wqkv  = (half_t*)(ws + p); p += (size_t)3 * Cc * Cc * 2;
  half_t* wproj = (half_t*)(ws + p); p += (size_t)Cc * Cc * 2;
  half_t* w1h   = (half_t*)(ws + p); p += (size_t)FFf * Cc * 2;
  half_t* w2h   = (half_t*)(ws + p); p += (size_t)Cc * FFf * 2;
  half_t* h1    = (half_t*)(ws + p); p += BNC * 2;
  half_t* qkv   = (half_t*)(ws + p); p += 3 * BNC * 2;   // q | k | v, each [B,H,N,D]
  half_t* aobuf = (half_t*)(ws + p); p += BNC * 2;
  float*  x1    = (float*)(ws + p);  p += BNC * 4;
  half_t* h2    = (half_t*)(ws + p); p += BNC * 2;
  half_t* mh    = (half_t*)(ws + p); p += (size_t)Mrows * FFf * 2;
  (void)p;

  cvt_f16<<<4096, 256, 0, stream>>>(qkv_w,  wqkv,  (size_t)3 * Cc * Cc);
  cvt_f16<<<4096, 256, 0, stream>>>(proj_w, wproj, (size_t)Cc * Cc);
  cvt_f16<<<4096, 256, 0, stream>>>(w1,     w1h,   (size_t)FFf * Cc);
  cvt_f16<<<4096, 256, 0, stream>>>(w2,     w2h,   (size_t)Cc * FFf);

  layernorm_f16<<<Mrows, 256, 0, stream>>>(x, ln1_g, ln1_b, h1);

  gemm_wmma<0><<<dim3(Mrows / 128, (3 * Cc) / 64), 256, 0, stream>>>(
      h1, wqkv, nullptr, nullptr, qkv, nullptr, Mrows, 3 * Cc, Cc);

  flash_attn<<<dim3(Nseq / 128, Bb * Hh), 256, 0, stream>>>(
      qkv, qkv + BNC, qkv + 2 * BNC, aobuf);

  gemm_wmma<2><<<dim3(Mrows / 128, Cc / 64), 256, 0, stream>>>(
      aobuf, wproj, proj_b, x, nullptr, x1, Mrows, Cc, Cc);

  layernorm_f16<<<Mrows, 256, 0, stream>>>(x1, ln2_g, ln2_b, h2);

  gemm_wmma<1><<<dim3(Mrows / 128, FFf / 64), 256, 0, stream>>>(
      h2, w1h, b1, nullptr, mh, nullptr, Mrows, FFf, Cc);

  gemm_wmma<2><<<dim3(Mrows / 128, Cc / 64), 256, 0, stream>>>(
      mh, w2h, b2, x1, nullptr, out, Mrows, Cc, FFf);
}
